// CausalSelfAttention_69793218560051
// MI455X (gfx1250) — compile-verified, matrix operand order checked by reading, not run on hardware
//
#include <hip/hip_runtime.h>

#define BB   4
#define SS   2048
#define DD   1024
#define HH   16
#define HDIM 64

typedef __attribute__((ext_vector_type(16))) __bf16       v16bf;
typedef __attribute__((ext_vector_type(8)))  float        v8f;
typedef __attribute__((ext_vector_type(4)))  unsigned int uiv4;
typedef __attribute__((ext_vector_type(8)))  int          iv8;
typedef __attribute__((ext_vector_type(4)))  int          iv4;

union FragBF { v16bf v; uiv4 q[2]; unsigned short s[16]; };
union AccF   { v8f   v; float f[8]; };
union Pack8  { uiv4  q; unsigned short s8[8]; };

__device__ __forceinline__ unsigned short f2bf(float x) {
  // native f32 -> bf16 convert (lowers to v_cvt_pk_bf16_f32 when paired)
  return __builtin_bit_cast(unsigned short, (__bf16)x);
}

// ---------------------------------------------------------------------------
// TDM: async 2D tile load Global->LDS.  tileW elements per row (data_size=2B),
// tileH rows, row stride strideElems elements.  D# per CDNA5 ISA 8.3/8.4.
// Tracked by TENSORcnt.
// ---------------------------------------------------------------------------
__device__ __forceinline__ void tdm_load_2d(unsigned long long gaddr,
                                            unsigned ldsOff, unsigned tileW,
                                            unsigned tileH, unsigned strideElems) {
  uiv4 g0;
  g0[0] = 1u;                                             // count=1, user D#
  g0[1] = ldsOff;                                         // lds_addr (bytes)
  g0[2] = (unsigned)(gaddr & 0xFFFFFFFFu);                // global_addr[31:0]
  g0[3] = (unsigned)((gaddr >> 32) & 0x01FFFFFFu) | (2u << 30);  // [56:32] | type=2
  iv8 g1;
  g1[0] = 0x10000;                 // workgroup_mask=0, data_size=1 (2 bytes)
  g1[1] = (int)(tileW << 16);      // tensor_dim0 (bits 79:48, low half)
  g1[2] = (int)(tileH << 16);      // tensor_dim1 (bits 111:80, low half)
  g1[3] = (int)(tileW << 16);      // tile_dim0   (bits 127:112)
  g1[4] = (int)tileH;              // tile_dim1, tile_dim2 = 0
  g1[5] = (int)strideElems;        // tensor_dim0_stride (elements)
  g1[6] = 0;
  g1[7] = 0;
  iv4 z4 = {0, 0, 0, 0};
#if __has_include(<hip/amd_detail/amd_gfx1250_TDM.h>)
  iv8 z8 = {0, 0, 0, 0, 0, 0, 0, 0};
  __builtin_amdgcn_tensor_load_to_lds(g0, g1, z4, z4, z8, 0);
#else
  __builtin_amdgcn_tensor_load_to_lds(g0, g1, z4, z4, 0);
#endif
}

// ---------------------------------------------------------------------------
// Kernel 1: QKV projection  C[8192,3072] = X[8192,1024] * Wqkv[1024,3072]
// 256 threads = 8 waves; workgroup tile 128(M) x 64(N); K step 32.
// Writes Q,K as bf16 [B,H,S,hd]; V as bf16 TRANSPOSED [B,H,hd,S] so the
// attention kernel can TDM the V^T tile directly (no LDS transpose).
// ---------------------------------------------------------------------------
__global__ __launch_bounds__(256) void qkv_proj_kernel(
    const float* __restrict__ X, const float* __restrict__ Wqkv,
    unsigned short* __restrict__ Q, unsigned short* __restrict__ K,
    unsigned short* __restrict__ Vt)
{
  __shared__ __align__(16) unsigned short ldsA[128 * 32];
  __shared__ __align__(16) unsigned short ldsB[64 * 32];
  const int t = threadIdx.x;
  const int lane = t & 31;
  const int w = t >> 5;
  const int rowBase = blockIdx.y * 128;
  const int colBase = blockIdx.x * 64;

  AccF acc[4];
#pragma unroll
  for (int nb = 0; nb < 4; ++nb)
#pragma unroll
    for (int i = 0; i < 8; ++i) acc[nb].f[i] = 0.0f;

  for (int k0 = 0; k0 < DD; k0 += 32) {
    // stage A: one row, 16 consecutive K per thread -> two b128 LDS stores
    {
      const int e0 = t * 16;
      const int r = e0 >> 5;
      const int kk0 = e0 & 31;
      const float* src = X + (size_t)(rowBase + r) * DD + k0 + kk0;
      if (k0 + 32 < DD) __builtin_prefetch(src + 32, 0, 3);
#pragma unroll
      for (int u = 0; u < 16; ++u) {
        const int kk = kk0 + u;
        const int hi = kk >> 4, rem = kk & 15;
        const int gg = rem >> 3, j = hi * 8 + (rem & 7);
        ldsA[(((r >> 4) * 32 + (r & 15) + 16 * gg) << 4) + j] = f2bf(src[u]);
      }
    }
    // stage B: fixed column n, 8 consecutive K per thread -> one b128 LDS store
    {
      const int n = t & 63;
      const int kk0 = (t >> 6) * 8;
      const float* src = Wqkv + (size_t)(k0 + kk0) * (3 * DD) + colBase + n;
      if (k0 + 32 < DD) __builtin_prefetch(src + (size_t)32 * (3 * DD), 0, 3);
      const int hi = kk0 >> 4;
      const int gg = (kk0 & 15) >> 3;
      Pack8 pk;
#pragma unroll
      for (int u = 0; u < 8; ++u) pk.s8[u] = f2bf(src[(size_t)u * (3 * DD)]);
      *(uiv4*)&ldsB[(((n >> 4) * 32 + (n & 15) + 16 * gg) << 4) + hi * 8] = pk.q;
    }
    __syncthreads();

    FragBF a;
    {
      const uiv4* pa = (const uiv4*)&ldsA[(w * 32 + lane) * 16];
      a.q[0] = pa[0]; a.q[1] = pa[1];
    }
#pragma unroll
    for (int nb = 0; nb < 4; ++nb) {
      FragBF b;
      const uiv4* pb = (const uiv4*)&ldsB[(nb * 32 + lane) * 16];
      b.q[0] = pb[0]; b.q[1] = pb[1];
      acc[nb].v = __builtin_amdgcn_wmma_f32_16x16x32_bf16(
          false, a.v, false, b.v, (short)0, acc[nb].v, false, false);
    }
    __syncthreads();
  }

  // scatter C tile: Q,K -> [B,H,S,hd] (scalar b16); V -> [B,H,hd,S]
  // where each lane's 8 rows are consecutive s -> one b128 store.
  const int half = lane >> 4, col = lane & 15;
#pragma unroll
  for (int nb = 0; nb < 4; ++nb) {
    const int n = colBase + nb * 16 + col;
    const int which = n >> 10;          // 0=Q 1=K 2=V
    const int dd = n & 1023;
    const int h = dd >> 6, hd_i = dd & 63;
    const int m0 = rowBase + w * 16 + half * 8;
    const int b = m0 >> 11, s0 = m0 & 2047;
    if (which == 2) {
      Pack8 pk;
#pragma unroll
      for (int i = 0; i < 8; ++i) pk.s8[i] = f2bf(acc[nb].f[i]);
      *(uiv4*)&Vt[((size_t)((b * HH + h) * HDIM + hd_i) << 11) + s0] = pk.q;
    } else {
      unsigned short* dst = (which == 0) ? Q : K;
#pragma unroll
      for (int i = 0; i < 8; ++i)
        dst[((size_t)((b * HH + h) * SS + s0 + i) << 6) + hd_i] = f2bf(acc[nb].f[i]);
    }
  }
}

// ---------------------------------------------------------------------------
// Kernel 2: causal flash attention. 128 threads = 4 waves.
// grid.x = S/64 q-blocks, grid.y = B*H. Wave w owns q rows [qblk*64+16w, +16).
// K (32x64, row-major) and V^T (64x32, strided rows of V^T global layout)
// tiles staged via TDM, DOUBLE BUFFERED with s_wait_tensorcnt(2).
// ---------------------------------------------------------------------------
__global__ __launch_bounds__(128) void attention_kernel(
    const unsigned short* __restrict__ Q, const unsigned short* __restrict__ K,
    const unsigned short* __restrict__ Vt, unsigned short* __restrict__ Y)
{
  __shared__ __align__(16) unsigned short ldsK[2][32 * 64];   // [buf][key][hd]
  __shared__ __align__(16) unsigned short ldsVT[2][64 * 32];  // [buf][hd][key]
  __shared__ __align__(16) unsigned short ldsP[4][16 * 32];

  const int t = threadIdx.x;
  const int lane = t & 31, w = t >> 5;
  const int g = lane >> 4, r = lane & 15;
  const int bh = blockIdx.y;
  const int qblk = blockIdx.x;
  const int qRow0 = qblk * 64 + w * 16;

  auto issue_tile = [&](int k0, int buf) {
    const unsigned long long gK =
        (unsigned long long)(const void*)(K + ((size_t)bh * SS + k0) * HDIM);
    const unsigned long long gV =
        (unsigned long long)(const void*)(Vt + ((size_t)bh * HDIM << 11) + k0);
    tdm_load_2d(gK, (unsigned)(unsigned long long)(void*)&ldsK[buf][0], 64, 32, 64);
    tdm_load_2d(gV, (unsigned)(unsigned long long)(void*)&ldsVT[buf][0], 32, 64, SS);
  };

  // Q fragments (two hd-halves), resident for the whole key loop
  FragBF qf[2];
  {
    const unsigned short* qrow = Q + ((size_t)bh * SS + (qRow0 + r)) * HDIM;
#pragma unroll
    for (int f = 0; f < 2; ++f) {
      qf[f].q[0] = *(const uiv4*)(qrow + 32 * f + 8 * g);
      qf[f].q[1] = *(const uiv4*)(qrow + 32 * f + 16 + 8 * g);
    }
  }

  float mrow[8], lrow[8];
  AccF o[4];
#pragma unroll
  for (int i = 0; i < 8; ++i) { mrow[i] = -3.0e38f; lrow[i] = 0.0f; }
#pragma unroll
  for (int nb = 0; nb < 4; ++nb)
#pragma unroll
    for (int i = 0; i < 8; ++i) o[nb].f[i] = 0.0f;

  const int nIter = 2 * qblk + 2;           // 32-key blocks up to causal bound
  if (w == 0) issue_tile(0, 0);             // prologue DMA

  for (int it = 0; it < nIter; ++it) {
    const int k0 = it * 32;
    const int cur = it & 1;

    if (w == 0) {
      if (it + 1 < nIter) {
        issue_tile(k0 + 32, cur ^ 1);           // prefetch next tile via TDM
        __builtin_amdgcn_s_wait_tensorcnt(2);   // tile 'it' complete, next in flight
      } else {
        __builtin_amdgcn_s_wait_tensorcnt(0);
      }
    }
    __syncthreads();

    // scores: two 16-key tiles, K-dim 64 = 2 chained WMMAs each
    AccF sc[2];
#pragma unroll
    for (int tt = 0; tt < 2; ++tt) {
      FragBF b0, b1;
      const unsigned short* krow = &ldsK[cur][(tt * 16 + r) * 64];
      b0.q[0] = *(const uiv4*)(krow + 8 * g);
      b0.q[1] = *(const uiv4*)(krow + 16 + 8 * g);
      b1.q[0] = *(const uiv4*)(krow + 32 + 8 * g);
      b1.q[1] = *(const uiv4*)(krow + 48 + 8 * g);
      AccF z;
#pragma unroll
      for (int i = 0; i < 8; ++i) z.f[i] = 0.0f;
      z.v = __builtin_amdgcn_wmma_f32_16x16x32_bf16(false, qf[0].v, false, b0.v, (short)0, z.v, false, false);
      z.v = __builtin_amdgcn_wmma_f32_16x16x32_bf16(false, qf[1].v, false, b1.v, (short)0, z.v, false, false);
      sc[tt] = z;
    }

    // causal mask + online softmax (row = qRow0 + 8g + i, col = k0 + 16tt + r)
#pragma unroll
    for (int i = 0; i < 8; ++i) {
      const int row = qRow0 + g * 8 + i;
      const float v0 = (k0 + r      <= row) ? sc[0].f[i] * 0.125f : -1.0e30f;
      const float v1 = (k0 + 16 + r <= row) ? sc[1].f[i] * 0.125f : -1.0e30f;
      float mx = fmaxf(v0, v1);
      mx = fmaxf(mx, __shfl_xor(mx, 1));
      mx = fmaxf(mx, __shfl_xor(mx, 2));
      mx = fmaxf(mx, __shfl_xor(mx, 4));
      mx = fmaxf(mx, __shfl_xor(mx, 8));
      const float mnew  = fmaxf(mrow[i], mx);
      const float alpha = __expf(mrow[i] - mnew);
      mrow[i] = mnew;
      const float p0 = __expf(v0 - mnew);
      const float p1 = __expf(v1 - mnew);
      float ls = p0 + p1;
      ls += __shfl_xor(ls, 1);
      ls += __shfl_xor(ls, 2);
      ls += __shfl_xor(ls, 4);
      ls += __shfl_xor(ls, 8);
      lrow[i] = lrow[i] * alpha + ls;
      sc[0].f[i] = p0; sc[1].f[i] = p1;
#pragma unroll
      for (int nb = 0; nb < 4; ++nb) o[nb].f[i] *= alpha;
    }

    // P -> LDS as bf16 [16 rows][32 keys] (per-wave buffer)
    {
      unsigned short* P = &ldsP[w][0];
#pragma unroll
      for (int i = 0; i < 8; ++i) {
        const int row = g * 8 + i;
        P[row * 32 + r]      = f2bf(sc[0].f[i]);
        P[row * 32 + 16 + r] = f2bf(sc[1].f[i]);
      }
    }
    __syncthreads();

    // O += P x V
    {
      const unsigned short* P = &ldsP[w][0];
      FragBF pf;
      pf.q[0] = *(const uiv4*)(P + r * 32 + 8 * g);
      pf.q[1] = *(const uiv4*)(P + r * 32 + 16 + 8 * g);
#pragma unroll
      for (int nb = 0; nb < 4; ++nb) {
        FragBF vf;
        const unsigned short* vcol = &ldsVT[cur][(nb * 16 + r) * 32];
        vf.q[0] = *(const uiv4*)(vcol + 8 * g);
        vf.q[1] = *(const uiv4*)(vcol + 16 + 8 * g);
        o[nb].v = __builtin_amdgcn_wmma_f32_16x16x32_bf16(
            false, pf.v, false, vf.v, (short)0, o[nb].v, false, false);
      }
    }
    __syncthreads();
  }

  // normalize and write Y bf16 in [B,S,D] with d = h*64 + hd
  const int b = bh >> 4, h = bh & 15;
#pragma unroll
  for (int nb = 0; nb < 4; ++nb) {
#pragma unroll
    for (int i = 0; i < 8; ++i) {
      const int row = qRow0 + g * 8 + i;
      const float yv = o[nb].f[i] / lrow[i];
      Y[((size_t)(b * SS + row) * DD) + h * HDIM + nb * 16 + r] = f2bf(yv);
    }
  }
}

// ---------------------------------------------------------------------------
// Kernel 3: output projection  Out[8192,1024] = Y[8192,1024] * Wout[1024,1024]
// Same tiling as kernel 1; A is already bf16; writes fp32 to d_out.
// ---------------------------------------------------------------------------
__global__ __launch_bounds__(256) void out_proj_kernel(
    const unsigned short* __restrict__ Ybf, const float* __restrict__ Wout,
    float* __restrict__ Out)
{
  __shared__ __align__(16) unsigned short ldsA[128 * 32];
  __shared__ __align__(16) unsigned short ldsB[64 * 32];
  const int t = threadIdx.x;
  const int lane = t & 31;
  const int w = t >> 5;
  const int rowBase = blockIdx.y * 128;
  const int colBase = blockIdx.x * 64;

  AccF acc[4];
#pragma unroll
  for (int nb = 0; nb < 4; ++nb)
#pragma unroll
    for (int i = 0; i < 8; ++i) acc[nb].f[i] = 0.0f;

  for (int k0 = 0; k0 < DD; k0 += 32) {
    {
      const int e0 = t * 16;
      const int r = e0 >> 5;
      const int kk0 = e0 & 31;
      const unsigned short* src = Ybf + (size_t)(rowBase + r) * DD + k0 + kk0;
      if (k0 + 32 < DD) __builtin_prefetch(src + 32, 0, 3);
#pragma unroll
      for (int u = 0; u < 16; ++u) {
        const int kk = kk0 + u;
        const int hi = kk >> 4, rem = kk & 15;
        const int gg = rem >> 3, j = hi * 8 + (rem & 7);
        ldsA[(((r >> 4) * 32 + (r & 15) + 16 * gg) << 4) + j] = src[u];
      }
    }
    {
      const int n = t & 63;
      const int kk0 = (t >> 6) * 8;
      const float* src = Wout + (size_t)(k0 + kk0) * DD + colBase + n;
      if (k0 + 32 < DD) __builtin_prefetch(src + (size_t)32 * DD, 0, 3);
      const int hi = kk0 >> 4;
      const int gg = (kk0 & 15) >> 3;
      Pack8 pk;
#pragma unroll
      for (int u = 0; u < 8; ++u) pk.s8[u] = f2bf(src[(size_t)u * DD]);
      *(uiv4*)&ldsB[(((n >> 4) * 32 + (n & 15) + 16 * gg) << 4) + hi * 8] = pk.q;
    }
    __syncthreads();

    FragBF a;
    {
      const uiv4* pa = (const uiv4*)&ldsA[(w * 32 + lane) * 16];
      a.q[0] = pa[0]; a.q[1] = pa[1];
    }
#pragma unroll
    for (int nb = 0; nb < 4; ++nb) {
      FragBF bfr;
      const uiv4* pb = (const uiv4*)&ldsB[(nb * 32 + lane) * 16];
      bfr.q[0] = pb[0]; bfr.q[1] = pb[1];
      acc[nb].v = __builtin_amdgcn_wmma_f32_16x16x32_bf16(
          false, a.v, false, bfr.v, (short)0, acc[nb].v, false, false);
    }
    __syncthreads();
  }

  const int half = lane >> 4, col = lane & 15;
#pragma unroll
  for (int nb = 0; nb < 4; ++nb) {
#pragma unroll
    for (int i = 0; i < 8; ++i) {
      const int m = rowBase + w * 16 + half * 8 + i;
      Out[(size_t)m * DD + colBase + nb * 16 + col] = acc[nb].f[i];
    }
  }
}

// ---------------------------------------------------------------------------
extern "C" void kernel_launch(void* const* d_in, const int* in_sizes, int n_in,
                              void* d_out, int out_size, void* d_ws, size_t ws_size,
                              hipStream_t stream) {
  (void)in_sizes; (void)n_in; (void)out_size; (void)ws_size;
  const float* x    = (const float*)d_in[0];   // [B,S,D]
  const float* wqkv = (const float*)d_in[1];   // [D,3D]
  const float* wout = (const float*)d_in[2];   // [D,D]
  float* out = (float*)d_out;                  // [B,S,D]

  const size_t elems = (size_t)BB * SS * DD;   // 8,388,608
  unsigned short* qbuf  = (unsigned short*)d_ws;
  unsigned short* kbuf  = qbuf + elems;
  unsigned short* vtbuf = kbuf + elems;        // V transposed [B,H,hd,S]
  unsigned short* ybuf  = vtbuf + elems;

  qkv_proj_kernel<<<dim3(48, 64), 256, 0, stream>>>(x, wqkv, qbuf, kbuf, vtbuf);
  attention_kernel<<<dim3(32, 64), 128, 0, stream>>>(qbuf, kbuf, vtbuf, ybuf);
  out_proj_kernel<<<dim3(16, 64), 256, 0, stream>>>(ybuf, wout, out);
}
